// LNNMotion_76270029242693
// MI455X (gfx1250) — matrix operand order repaired and checked by reading, not compiled, not run on hardware
//
#include <hip/hip_runtime.h>
#include <hip/hip_bf16.h>

typedef __attribute__((ext_vector_type(16))) __bf16 v16bf;
typedef __attribute__((ext_vector_type(8)))  float  v8f;

#define B_TOT 1024
#define T_LEN 128
#define IN_D  2
#define H_D   256
#define OUT_D 2

#define TILE_M  16            // batch rows per workgroup (WMMA M)
#define WAVES   16            // one wave per 16-wide N tile -> covers H=256
#define THREADS (WAVES * 32)

union Frag16 { unsigned int u[8]; v16bf v; };

__device__ __forceinline__ unsigned short f32_to_bf16(float f) {
    unsigned int u = __float_as_uint(f);
    u += 0x7FFFu + ((u >> 16) & 1u);      // round-to-nearest-even
    return (unsigned short)(u >> 16);
}

// Hardware tanh: gfx1250 has V_TANH_F32 (TRANS pipe, co-executes with WMMA).
__device__ __forceinline__ float fast_tanh(float x) {
#if __has_builtin(__builtin_amdgcn_tanhf)
    return __builtin_amdgcn_tanhf(x);
#else
    float y;
    // v_nop's cover the TRANS-result hazard since the compiler can't see
    // through the asm that this is a transcendental op.
    asm volatile("v_tanh_f32 %0, %1\n\tv_nop\n\tv_nop"
                 : "=v"(y) : "v"(x));
    return y;
#endif
}

__global__ __launch_bounds__(THREADS)
void lnn_persistent_kernel(const float* __restrict__ x,
                           const float* __restrict__ w_in,
                           const float* __restrict__ b_in,
                           const float* __restrict__ w_h,
                           const float* __restrict__ b_h,
                           const float* __restrict__ alpha,
                           const float* __restrict__ beta,
                           const float* __restrict__ fc_w,
                           const float* __restrict__ fc_b,
                           float* __restrict__ out)
{
    // h tile as bf16, ping-pong double buffer: [buf][m][k-pair] packed 2 bf16/dword
    __shared__ unsigned int lds_h16[2][TILE_M * (H_D / 2)];
    // staged input tile: [m][t][2]
    __shared__ float lds_x[TILE_M * T_LEN * IN_D];
    // final f32 h for the output reduction
    __shared__ float lds_hf[TILE_M * H_D];

    const int tid  = threadIdx.x;
    const int wave = tid >> 5;
    const int lane = tid & 31;
    const int b0   = blockIdx.x * TILE_M;

    // ---- stage x tile (coalesced, same [m][t][i] layout as global) ----
    for (int i = tid; i < TILE_M * T_LEN * IN_D; i += THREADS)
        lds_x[i] = x[b0 * (T_LEN * IN_D) + i];

    // ---- h0 = 0 ----
    for (int i = tid; i < TILE_M * (H_D / 2); i += THREADS)
        lds_h16[0][i] = 0u;

    // ---- per-lane constants (n fixed for the whole kernel) ----
    const int nloc = lane & 15;
    const int half = lane >> 4;                 // selects M-half (C/D) and K-half (A/B)
    const int n    = wave * 16 + nloc;          // hidden unit owned by this lane column
    const float wi0 = w_in[n * IN_D + 0];
    const float wi1 = w_in[n * IN_D + 1];
    const float cb  = b_in[n] + b_h[n];
    const float g   = alpha[n] * beta[n];       // h' = h + g*(tanh(pre) - h)

    // ---- build resident bf16 B-fragments: B[k][n] = w_h[n][k]  (W_h^T panel) ----
    // B in this code's contiguous k order (same as A): lane = half*16 + n%16 holds K rows half*16..+15,
    // VGPR r packs (k0, k0+1) with k0 = kk*32 + half*16 + 2r.
    Frag16 bfr[8];
    #pragma unroll
    for (int kk = 0; kk < 8; ++kk) {
        #pragma unroll
        for (int r = 0; r < 8; ++r) {
            const int k = kk * 32 + half * 16 + 2 * r;
            const unsigned int lo = f32_to_bf16(w_h[n * H_D + k]);
            const unsigned int hi = f32_to_bf16(w_h[n * H_D + k + 1]);
            bfr[kk].u[r] = lo | (hi << 16);
        }
    }

    // f32 h state in registers: lane owns (m, n) with m = r + 8*half (C/D layout)
    float hreg[8];
    #pragma unroll
    for (int r = 0; r < 8; ++r) hreg[r] = 0.f;

    __syncthreads();

    int p = 0;                                   // current read buffer
    for (int t = 0; t < T_LEN; ++t) {
        // ---- GEMM: pre_tile = h_tile(16xK) @ W_h^T panel (Kx16), K=256 ----
        v8f acc = {};
        #pragma unroll
        for (int kk = 0; kk < 8; ++kk) {
            // A in this code's contiguous k order (same as B): m = lane%16; lanes 0-15 K=0..15, lanes 16-31 K=16..31
            // VGPR0-3: K = kb+0..7 ; VGPR4-7: K = kb+8..15  (kb = 16*half)
            Frag16 afr;
            const unsigned int* row = &lds_h16[p][(lane & 15) * (H_D / 2)];
            const int base = kk * 16 + half * 8;   // dword index of K-pair
            #pragma unroll
            for (int q = 0; q < 4; ++q) afr.u[q]     = row[base + q];
            #pragma unroll
            for (int q = 0; q < 4; ++q) afr.u[4 + q] = row[base + 4 + q];

            acc = __builtin_amdgcn_wmma_f32_16x16x32_bf16(
                      false, afr.v, false, bfr[kk].v,
                      (short)0, acc, false, false);
        }

        // ---- elementwise liquid update; write new h (bf16) to the other buffer ----
        unsigned short* wrow = (unsigned short*)lds_h16[1 - p];
        #pragma unroll
        for (int r = 0; r < 8; ++r) {
            const int m  = r + 8 * half;
            const float x0 = lds_x[m * (T_LEN * IN_D) + t * IN_D + 0];
            const float x1 = lds_x[m * (T_LEN * IN_D) + t * IN_D + 1];
            const float pre = acc[r] + x0 * wi0 + x1 * wi1 + cb;
            const float hn  = hreg[r] + g * (fast_tanh(pre) - hreg[r]);
            hreg[r] = hn;
            wrow[m * H_D + n] = f32_to_bf16(hn);
        }
        p ^= 1;
        __syncthreads();
    }

    // ---- dump f32 h and do out = h @ fc_w^T + fc_b (OUT=2) ----
    #pragma unroll
    for (int r = 0; r < 8; ++r)
        lds_hf[(r + 8 * half) * H_D + n] = hreg[r];
    __syncthreads();

    {
        const int m = wave;                      // wave w reduces batch row m=w
        float s0 = 0.f, s1 = 0.f;
        #pragma unroll
        for (int j = lane; j < H_D; j += 32) {
            const float hv = lds_hf[m * H_D + j];
            s0 += hv * fc_w[0 * H_D + j];
            s1 += hv * fc_w[1 * H_D + j];
        }
        #pragma unroll
        for (int off = 16; off > 0; off >>= 1) {
            s0 += __shfl_down(s0, off, 32);
            s1 += __shfl_down(s1, off, 32);
        }
        if (lane == 0) {
            out[(b0 + m) * OUT_D + 0] = s0 + fc_b[0];
            out[(b0 + m) * OUT_D + 1] = s1 + fc_b[1];
        }
    }
}

extern "C" void kernel_launch(void* const* d_in, const int* in_sizes, int n_in,
                              void* d_out, int out_size, void* d_ws, size_t ws_size,
                              hipStream_t stream) {
    (void)in_sizes; (void)n_in; (void)out_size; (void)d_ws; (void)ws_size;
    const float* x     = (const float*)d_in[0];
    const float* w_in  = (const float*)d_in[1];
    const float* b_in  = (const float*)d_in[2];
    const float* w_h   = (const float*)d_in[3];
    const float* b_h   = (const float*)d_in[4];
    const float* alpha = (const float*)d_in[5];
    const float* beta  = (const float*)d_in[6];
    const float* fc_w  = (const float*)d_in[7];
    const float* fc_b  = (const float*)d_in[8];
    float* out = (float*)d_out;

    dim3 grid(B_TOT / TILE_M);      // 64 independent batch tiles
    dim3 block(THREADS);            // 16 waves (wave32)
    lnn_persistent_kernel<<<grid, block, 0, stream>>>(
        x, w_in, b_in, w_h, b_h, alpha, beta, fc_w, fc_b, out);
}
